// custom_gnn_90933047591155
// MI455X (gfx1250) — matrix operand order repaired and checked by reading, not compiled, not run on hardware
//
#include <hip/hip_runtime.h>
#include <hip/hip_bf16.h>
#include <stddef.h>

#define NNODES 50000
#define NEDGES 200000
#define DFEAT  384
#define NREL   2

#define BM 128   // block rows
#define BN 128   // block cols
#define KT 32    // k tile (WMMA K for bf16)
#define NKT (DFEAT / KT)   // 12 k-steps

typedef __attribute__((ext_vector_type(16))) __bf16 v16bf;
typedef __attribute__((ext_vector_type(8)))  float  v8f;

union FragAB {
  v16bf v;
  uint4 q[2];
};

static __device__ __forceinline__ unsigned int f2bf(float f) {
  // float -> bf16 round-to-nearest-even (NaN payload edge cases ignored)
  unsigned int u = __float_as_uint(f);
  unsigned int lsb = (u >> 16) & 1u;
  return (u + 0x7FFFu + lsb) >> 16;
}

struct StageRegs {
  float4 a[4];   // A tile slice: BM x KT floats, 4 float4 per thread
  float4 b[4];   // B tile slice: KT x BN floats, 4 float4 per thread
};

static __device__ __forceinline__ void stage_load(
    const float* __restrict__ X, const float* __restrict__ Wt, int nrows,
    int row0, int col0, int k0, int tid, StageRegs& s) {
#pragma unroll
  for (int i = 0; i < 4; ++i) {           // A: [row][k] row-major global
    int f4 = tid + i * 256;               // 0..1023
    int r  = f4 >> 3;                     // 8 float4 per row of KT
    int kk = (f4 & 7) * 4;
    int gr = row0 + r;
    s.a[i] = make_float4(0.f, 0.f, 0.f, 0.f);
    if (gr < nrows) s.a[i] = *(const float4*)(X + (size_t)gr * DFEAT + k0 + kk);
  }
#pragma unroll
  for (int i = 0; i < 4; ++i) {           // B: [k][n] row-major global
    int f4 = tid + i * 256;
    int k  = f4 >> 5;                     // 32 float4 per k-row of BN
    int nn = (f4 & 31) * 4;
    s.b[i] = *(const float4*)(Wt + (size_t)(k0 + k) * DFEAT + col0 + nn);
  }
}

static __device__ __forceinline__ void stage_store(
    unsigned short* ldsA, unsigned short* ldsB, int tid, const StageRegs& s) {
#pragma unroll
  for (int i = 0; i < 4; ++i) {           // A -> [row][k] bf16
    int f4 = tid + i * 256;
    int r  = f4 >> 3;
    int kk = (f4 & 7) * 4;
    uint2 p;
    p.x = f2bf(s.a[i].x) | (f2bf(s.a[i].y) << 16);
    p.y = f2bf(s.a[i].z) | (f2bf(s.a[i].w) << 16);
    *(uint2*)(&ldsA[r * KT + kk]) = p;
  }
#pragma unroll
  for (int i = 0; i < 4; ++i) {           // B -> transposed [n][k] bf16
    int f4 = tid + i * 256;
    int k  = f4 >> 5;
    int nn = (f4 & 31) * 4;
    ldsB[(nn + 0) * KT + k] = (unsigned short)f2bf(s.b[i].x);
    ldsB[(nn + 1) * KT + k] = (unsigned short)f2bf(s.b[i].y);
    ldsB[(nn + 2) * KT + k] = (unsigned short)f2bf(s.b[i].z);
    ldsB[(nn + 3) * KT + k] = (unsigned short)f2bf(s.b[i].w);
  }
}

// ---------------------------------------------------------------------------
// GEMM: Out[m,n] = sum_k X[m,k] * Wt[k,n]  (+ bias[n] if bias != nullptr)
// bf16 WMMA, f32 accumulate; double-buffered LDS, register prefetch pipeline.
// grid = (ceil(nrows/BM), DFEAT/BN), block = 256 (8 waves)
// ---------------------------------------------------------------------------
__global__ __launch_bounds__(256) void rgcn_gemm_bf16(
    const float* __restrict__ X, const float* __restrict__ Wt,
    const float* __restrict__ bias, float* __restrict__ Out, int nrows)
{
  __shared__ unsigned short ldsA[2][BM * KT];  // [row][k] bf16
  __shared__ unsigned short ldsB[2][BN * KT];  // [n][k]  bf16 (transposed)

  const int tid  = threadIdx.x;
  const int lane = tid & 31;
  const int wave = tid >> 5;          // 0..7 -> row sub-tile
  const int half = (lane >> 4) & 1;   // wave32 half
  const int l16  = lane & 15;

  const int row0 = blockIdx.x * BM;
  const int col0 = blockIdx.y * BN;

  v8f zero = {};
  v8f acc[BN / 16];
#pragma unroll
  for (int t = 0; t < BN / 16; ++t) acc[t] = zero;

  StageRegs sr;
  stage_load(X, Wt, nrows, row0, col0, 0, tid, sr);
  stage_store(ldsA[0], ldsB[0], tid, sr);
  __syncthreads();

  int cur = 0;
  for (int kt = 0; kt < NKT; ++kt) {
    // prefetch next k-tile from global while computing current one
    if (kt + 1 < NKT)
      stage_load(X, Wt, nrows, row0, col0, (kt + 1) * KT, tid, sr);

    // A fragment: 16x32, lanes 0-15 hold K=0..7,16..23; lanes 16-31 K=8..15,24..31
    FragAB aF;
    {
      int base = (wave * 16 + l16) * KT + 8 * half;
      aF.q[0] = *(const uint4*)(&ldsA[cur][base]);
      aF.q[1] = *(const uint4*)(&ldsA[cur][base + 16]);
    }
    // Batch-load all 8 B fragments (keep them live), then the 8 WMMAs
    FragAB bF[BN / 16];
#pragma unroll
    for (int nt = 0; nt < BN / 16; ++nt) {
      int base = (nt * 16 + l16) * KT + 8 * half;
      bF[nt].q[0] = *(const uint4*)(&ldsB[cur][base]);
      bF[nt].q[1] = *(const uint4*)(&ldsB[cur][base + 16]);
    }
    // Fence the scheduler: don't sink the ds_loads into the WMMA chain.
    // One s_wait_dscnt then 8 back-to-back v_wmma (dense matrix-pipe window).
    __builtin_amdgcn_sched_barrier(0);
#pragma unroll
    for (int nt = 0; nt < BN / 16; ++nt)
      acc[nt] = __builtin_amdgcn_wmma_f32_16x16x32_bf16(
          false, aF.v, false, bF[nt].v, (short)0, acc[nt], false, false);

    if (kt + 1 < NKT) {
      stage_store(ldsA[cur ^ 1], ldsB[cur ^ 1], tid, sr);
      __syncthreads();   // single barrier per iteration (double-buffered)
      cur ^= 1;
    }
  }

  // ---- epilogue: C/D layout lane -> N=l16, VGPR v -> M=v+8*half ----
#pragma unroll
  for (int nt = 0; nt < BN / 16; ++nt) {
    int n = col0 + nt * 16 + l16;
    float bv = (bias != nullptr) ? bias[n] : 0.0f;
#pragma unroll
    for (int v = 0; v < 8; ++v) {
      int r = row0 + wave * 16 + v + 8 * half;
      if (r < nrows) Out[(size_t)r * DFEAT + n] = acc[nt][v] + bv;
    }
  }
}

// ---------------------------------------------------------------------------
// Helper kernels
// ---------------------------------------------------------------------------
__global__ void fill_zero(float* p, int n) {
  int i = blockIdx.x * blockDim.x + threadIdx.x;
  if (i < n) p[i] = 0.0f;
}

__global__ void count_edges(const int* __restrict__ dst,
                            const int* __restrict__ etype,
                            float* __restrict__ cnt, int nE) {
  int e = blockIdx.x * blockDim.x + threadIdx.x;
  if (e >= nE) return;
  int r = etype[e];
  atomicAdd(cnt + (size_t)r * NNODES + dst[e], 1.0f);
}

__global__ void invert_counts(float* __restrict__ cnt, int n) {
  int i = blockIdx.x * blockDim.x + threadIdx.x;
  if (i < n) cnt[i] = 1.0f / fmaxf(cnt[i], 1.0f);
}

// one block (96 threads) per edge; adds xr[src] * invcnt[rel][dst] into acc[dst]
__global__ void scatter_mean(const float* __restrict__ xr,
                             const int* __restrict__ src,
                             const int* __restrict__ dst,
                             const int* __restrict__ etype,
                             const float* __restrict__ invc,
                             float* __restrict__ acc, int rel, int nE) {
  int e = blockIdx.x;
  if (e >= nE) return;
  if (etype[e] != rel) return;
  int s = src[e], d = dst[e];
  float scale = invc[(size_t)rel * NNODES + d];
  const float* xs = xr + (size_t)s * DFEAT;
  float* ad = acc + (size_t)d * DFEAT;
  int j = threadIdx.x * 4;                      // 96 threads * 4 = 384
  float4 xv = *(const float4*)(xs + j);
  atomicAdd(ad + j + 0, xv.x * scale);
  atomicAdd(ad + j + 1, xv.y * scale);
  atomicAdd(ad + j + 2, xv.z * scale);
  atomicAdd(ad + j + 3, xv.w * scale);
}

__global__ void relu_inplace(float* __restrict__ p, int n) {
  int i = blockIdx.x * blockDim.x + threadIdx.x;
  if (i < n) p[i] = fmaxf(p[i], 0.0f);
}

// ---------------------------------------------------------------------------
// Host orchestration
// ---------------------------------------------------------------------------
extern "C" void kernel_launch(void* const* d_in, const int* in_sizes, int n_in,
                              void* d_out, int out_size, void* d_ws, size_t ws_size,
                              hipStream_t stream) {
  (void)in_sizes; (void)n_in; (void)out_size; (void)ws_size;

  const float* emb   = (const float*)d_in[0];
  const int*   eidx  = (const int*)d_in[1];   // [2, E]: src row then dst row
  const int*   etype = (const int*)d_in[2];
  const float* W1    = (const float*)d_in[3];
  const float* root1 = (const float*)d_in[4];
  const float* bias1 = (const float*)d_in[5];
  const float* W2    = (const float*)d_in[6];
  const float* root2 = (const float*)d_in[7];
  const float* bias2 = (const float*)d_in[8];
  const float* W3    = (const float*)d_in[9];
  const float* root3 = (const float*)d_in[10];
  const float* bias3 = (const float*)d_in[11];

  const int* src = eidx;
  const int* dst = eidx + NEDGES;

  const size_t nd = (size_t)NNODES * DFEAT;
  float* bufA = (float*)d_ws;          // layer-1 output
  float* bufB = bufA + nd;             // layer-2 output
  float* XR   = bufB + nd;             // per-relation transformed features
  float* CNT  = XR + nd;               // [NREL, NNODES] counts -> inverse counts

  // per-(relation,dst) inverse mean counts (recomputed every call)
  fill_zero<<<(NREL * NNODES + 255) / 256, 256, 0, stream>>>(CNT, NREL * NNODES);
  count_edges<<<(NEDGES + 255) / 256, 256, 0, stream>>>(dst, etype, CNT, NEDGES);
  invert_counts<<<(NREL * NNODES + 255) / 256, 256, 0, stream>>>(CNT, NREL * NNODES);

  dim3 ggrid((NNODES + BM - 1) / BM, DFEAT / BN);

  auto layer = [&](const float* xin, float* xout, const float* W,
                   const float* root, const float* bias, bool do_relu) {
    // xout = xin @ root + bias (fully overwrites xout)
    rgcn_gemm_bf16<<<ggrid, 256, 0, stream>>>(xin, root, bias, xout, NNODES);
    for (int r = 0; r < NREL; ++r) {
      // XR = xin @ W[r]
      rgcn_gemm_bf16<<<ggrid, 256, 0, stream>>>(
          xin, W + (size_t)r * DFEAT * DFEAT, nullptr, XR, NNODES);
      // xout[dst] += XR[src] * invcnt[r][dst]   (mean aggregation)
      scatter_mean<<<NEDGES, 96, 0, stream>>>(XR, src, dst, etype, CNT, xout, r, NEDGES);
    }
    if (do_relu)
      relu_inplace<<<(int)((nd + 255) / 256), 256, 0, stream>>>(xout, (int)nd);
  };

  layer(emb,  bufA, W1, root1, bias1, true);
  layer(bufA, bufB, W2, root2, bias2, true);
  layer(bufB, (float*)d_out, W3, root3, bias3, false);
}
